// EGNNLayer_82772609728932
// MI455X (gfx1250) — matrix operand order repaired and checked by reading, not compiled, not run on hardware
//
#include <hip/hip_runtime.h>
#include <hip/hip_bf16.h>
#include <cstdint>
#include <cstddef>

typedef __bf16 bf16_t;
typedef __attribute__((ext_vector_type(16))) __bf16 v16bf;
typedef __attribute__((ext_vector_type(8)))  __bf16 v8bf;
typedef __attribute__((ext_vector_type(8)))  float  v8f;

#define NN   20000
#define NE   640000
#define NH   128
#define NED  16
#define EIN_S 288   // e_in K padded 273 -> 288, also LDS row stride (bf16)
#define MT_S  136   // 128 + 8 pad
#define CAT_S 264   // 256 + 8 pad

// ---------------- helpers ----------------

__device__ __forceinline__ float silu_f(float v) { return v / (1.0f + __expf(-v)); }

__device__ __forceinline__ v8f splat8(float b) {
    v8f r;
#pragma unroll
    for (int i = 0; i < 8; ++i) r[i] = b;
    return r;
}

// Build A operand (16x32 bf16) from an LDS row pointer, per ISA 16-bit A layout:
// lane holds row M = lane%16; VGPR0..3 -> K = kt*32 + hi*8 + [0..7],
// VGPR4..7 -> K = kt*32 + 16 + hi*8 + [0..7], hi = lane/16.
__device__ __forceinline__ v16bf load_A(const bf16_t* rowp, int kt, int hi) {
    const int k0 = kt * 32 + hi * 8;
    union { v16bf v; v8bf h[2]; } u;
    u.h[0] = *(const v8bf*)(rowp + k0);
    u.h[1] = *(const v8bf*)(rowp + k0 + 16);
    return u.v;
}

// One 16-row x 128-col MLP layer on a wave: acc[t] = bias + tile @ Wp (8 N-tiles).
__device__ __forceinline__ void mlp8(const bf16_t* rowp, int KT,
                                     const bf16_t* __restrict__ Wp,
                                     const float* __restrict__ bias,
                                     int lane, v8f acc[8]) {
    const int l16 = lane & 15;
    const int hi  = lane >> 4;
#pragma unroll
    for (int t = 0; t < 8; ++t) acc[t] = splat8(bias[t * 16 + l16]);
    for (int kt = 0; kt < KT; ++kt) {
        v16bf A = load_A(rowp, kt, hi);
        const bf16_t* bp = Wp + (size_t)kt * 8 * 512 + (size_t)lane * 16;
#pragma unroll
        for (int t = 0; t < 8; ++t) {
            v16bf B = *(const v16bf*)(bp + (size_t)t * 512);
            acc[t] = __builtin_amdgcn_wmma_f32_16x16x32_bf16(
                false, A, false, B, (short)0, acc[t], false, false);
        }
    }
}

// ---------------- utility kernels ----------------

__global__ void zero_f32(float* p, int n) {
    int i = blockIdx.x * blockDim.x + threadIdx.x;
    if (i < n) p[i] = 0.0f;
}

// Pack row-major f32 W[K][N] into CDNA5 WMMA B-layout bf16 tiles:
// dst[((kt*NT+nt)*32 + lane)*16 + e] = W[kt*32 + (lane/16)*8 + (e%8) + (e/8)*16][nt*16 + lane%16], e = 2v + p
__global__ void pack_b(const float* __restrict__ W, bf16_t* __restrict__ dst,
                       int K, int N, int KT, int NT) {
    int total = KT * NT * 512;
    for (int i = blockIdx.x * blockDim.x + threadIdx.x; i < total;
         i += gridDim.x * blockDim.x) {
        int p    = i & 1;
        int v    = (i >> 1) & 7;
        int lane = (i >> 4) & 31;
        int tile = i >> 9;
        int nt = tile % NT, kt = tile / NT;
        int col = nt * 16 + (lane & 15);
        int k   = kt * 32 + (lane >> 4) * 8 + ((2 * v + p) & 7) + (((2 * v + p) >> 3) << 4);
        float val = (k < K && col < N) ? W[(size_t)k * N + col] : 0.0f;
        dst[i] = (bf16_t)val;
    }
}

// ---------------- edge kernel ----------------

struct EdgeLDS {
    bf16_t ein[16 * EIN_S];   // e_in tile, bf16
    bf16_t mt[16 * MT_S];     // reused: silu(L1) then m_ij
    float  red[32 * 8];       // Wx2 reduction scratch
    int    ssrc[16];
    int    sdst[16];
};

__global__ __launch_bounds__(128, 1) void edge_kernel(
    const float* __restrict__ h, const float* __restrict__ x,
    const int* __restrict__ src, const int* __restrict__ dst,
    const float* __restrict__ edge_attr,
    const bf16_t* __restrict__ We1p, const float* __restrict__ be1,
    const bf16_t* __restrict__ We2p, const float* __restrict__ be2,
    const bf16_t* __restrict__ Wx1p, const float* __restrict__ bx1,
    const float* __restrict__ Wx2, const float* __restrict__ bx2,
    float* __restrict__ msg_agg, float* __restrict__ coord_agg) {
    __shared__ EdgeLDS sm[4];
    const int lane = threadIdx.x & 31;
    const int wave = threadIdx.x >> 5;
    EdgeLDS& S = sm[wave];
    const int ebase = (blockIdx.x * 4 + wave) * 16;
    const int l16 = lane & 15;
    const int hi  = lane >> 4;

    float dx = 0.f, dy = 0.f, dz = 0.f;
    if (lane < 16) {
        int e = ebase + lane;
        int s_ = src[e], d_ = dst[e];
        S.ssrc[lane] = s_;
        S.sdst[lane] = d_;
        dx = x[s_ * 3 + 0] - x[d_ * 3 + 0];
        dy = x[s_ * 3 + 1] - x[d_ * 3 + 1];
        dz = x[s_ * 3 + 2] - x[d_ * 3 + 2];
        float dsn = (dx * dx + dy * dy + dz * dz) * 0.01f;
        S.ein[lane * EIN_S + 256] = (bf16_t)dsn;
        for (int c = 273; c < 288; ++c) S.ein[lane * EIN_S + c] = (bf16_t)0.0f;
    }
    __syncthreads();

    // fill cols 0..255 (h[src]|h[dst]) and 257..272 (edge_attr)
    for (int idx = lane; idx < 16 * 272; idx += 32) {
        int row = idx / 272;
        int c   = idx % 272;
        int col = (c < 256) ? c : c + 1;  // skip dsn col 256
        float val;
        if (col < 128)        val = h[(size_t)S.ssrc[row] * NH + col];
        else if (col < 256)   val = h[(size_t)S.sdst[row] * NH + (col - 128)];
        else                  val = edge_attr[(size_t)(ebase + row) * NED + (col - 257)];
        S.ein[row * EIN_S + col] = (bf16_t)val;
    }
    __syncthreads();

    v8f acc[8];

    // L1: e_in (K=288 padded) @ We1 + be1 -> silu -> LDS
    mlp8(&S.ein[l16 * EIN_S], 9, We1p, be1, lane, acc);
#pragma unroll
    for (int t = 0; t < 8; ++t)
#pragma unroll
        for (int v = 0; v < 8; ++v) {
            int row = v + 8 * hi;
            S.mt[row * MT_S + t * 16 + l16] = (bf16_t)silu_f(acc[t][v]);
        }
    __syncthreads();

    // L2: @ We2 + be2 -> silu = m_ij ; scatter to msg_agg; stage bf16 for L3
    mlp8(&S.mt[l16 * MT_S], 4, We2p, be2, lane, acc);
#pragma unroll
    for (int t = 0; t < 8; ++t)
#pragma unroll
        for (int v = 0; v < 8; ++v) {
            float val = silu_f(acc[t][v]);
            int row = v + 8 * hi;
            int node = S.sdst[row];
            atomicAdd(&msg_agg[(size_t)node * NH + t * 16 + l16], val);
            S.mt[row * MT_S + t * 16 + l16] = (bf16_t)val;
        }
    __syncthreads();

    // L3: m_ij @ Wx1 + bx1 -> silu -> dot Wx2 (VALU + LDS reduce) -> tanh
    mlp8(&S.mt[l16 * MT_S], 4, Wx1p, bx1, lane, acc);
    float part[8];
#pragma unroll
    for (int v = 0; v < 8; ++v) part[v] = 0.0f;
#pragma unroll
    for (int t = 0; t < 8; ++t) {
        float w2 = Wx2[t * 16 + l16];
#pragma unroll
        for (int v = 0; v < 8; ++v) part[v] += silu_f(acc[t][v]) * w2;
    }
#pragma unroll
    for (int v = 0; v < 8; ++v) S.red[lane * 8 + v] = part[v];
    __syncthreads();

    if (lane < 16) {
        int row = lane;
        int v = row & 7, h2 = row >> 3;
        float sum = 0.0f;
        for (int l = 0; l < 16; ++l) sum += S.red[(h2 * 16 + l) * 8 + v];
        float cw = tanhf(sum + bx2[0]);
        int node = S.sdst[row];
        atomicAdd(&coord_agg[node * 3 + 0], dx * cw);
        atomicAdd(&coord_agg[node * 3 + 1], dy * cw);
        atomicAdd(&coord_agg[node * 3 + 2], dz * cw);
    }
}

// ---------------- node kernel ----------------

struct NodeLDS {
    bf16_t cat[16 * CAT_S];
    bf16_t t1[16 * MT_S];
};

__global__ __launch_bounds__(64, 1) void node_kernel(
    const float* __restrict__ h, const float* __restrict__ x,
    const unsigned char* __restrict__ fixed_mask,
    const bf16_t* __restrict__ Wh1p, const float* __restrict__ bh1,
    const bf16_t* __restrict__ Wh2p, const float* __restrict__ bh2,
    const float* __restrict__ msg_agg, const float* __restrict__ coord_agg,
    float* __restrict__ out_h, float* __restrict__ out_x) {
    __shared__ NodeLDS sm[2];
    const int lane = threadIdx.x & 31;
    const int wave = threadIdx.x >> 5;
    NodeLDS& S = sm[wave];
    const int nbase = (blockIdx.x * 2 + wave) * 16;
    const int l16 = lane & 15;
    const int hi  = lane >> 4;

    for (int idx = lane; idx < 16 * 256; idx += 32) {
        int row = idx >> 8;
        int col = idx & 255;
        size_t node = nbase + row;
        float val = (col < 128) ? h[node * NH + col]
                                : msg_agg[node * NH + (col - 128)];
        S.cat[row * CAT_S + col] = (bf16_t)val;
    }
    __syncthreads();

    v8f acc[8];
    mlp8(&S.cat[l16 * CAT_S], 8, Wh1p, bh1, lane, acc);
#pragma unroll
    for (int t = 0; t < 8; ++t)
#pragma unroll
        for (int v = 0; v < 8; ++v) {
            int row = v + 8 * hi;
            S.t1[row * MT_S + t * 16 + l16] = (bf16_t)silu_f(acc[t][v]);
        }
    __syncthreads();

    mlp8(&S.t1[l16 * MT_S], 4, Wh2p, bh2, lane, acc);
#pragma unroll
    for (int t = 0; t < 8; ++t)
#pragma unroll
        for (int v = 0; v < 8; ++v) {
            int row = v + 8 * hi;
            size_t node = nbase + row;
            int col = t * 16 + l16;
            out_h[node * NH + col] = h[node * NH + col] + acc[t][v];
        }

    if (lane < 16) {
        size_t node = nbase + lane;
        float m = fixed_mask[node] ? 0.0f : 1.0f;
        out_x[node * 3 + 0] = x[node * 3 + 0] + coord_agg[node * 3 + 0] * m;
        out_x[node * 3 + 1] = x[node * 3 + 1] + coord_agg[node * 3 + 1] * m;
        out_x[node * 3 + 2] = x[node * 3 + 2] + coord_agg[node * 3 + 2] * m;
    }
}

// ---------------- launch ----------------

extern "C" void kernel_launch(void* const* d_in, const int* in_sizes, int n_in,
                              void* d_out, int out_size, void* d_ws, size_t ws_size,
                              hipStream_t stream) {
    (void)in_sizes; (void)n_in; (void)out_size; (void)ws_size;
    const float* h    = (const float*)d_in[0];
    const float* x    = (const float*)d_in[1];
    const int*   ei   = (const int*)d_in[2];
    const float* ea   = (const float*)d_in[3];
    const unsigned char* fm = (const unsigned char*)d_in[4];
    const float* We1 = (const float*)d_in[5];
    const float* be1 = (const float*)d_in[6];
    const float* We2 = (const float*)d_in[7];
    const float* be2 = (const float*)d_in[8];
    const float* Wx1 = (const float*)d_in[9];
    const float* bx1 = (const float*)d_in[10];
    const float* Wx2 = (const float*)d_in[11];
    const float* bx2 = (const float*)d_in[12];
    const float* Wh1 = (const float*)d_in[13];
    const float* bh1 = (const float*)d_in[14];
    const float* Wh2 = (const float*)d_in[15];
    const float* bh2 = (const float*)d_in[16];

    const int* src = ei;            // edge_index[0]
    const int* dst = ei + NE;       // edge_index[1]

    // workspace layout
    float* msg_agg   = (float*)d_ws;                    // N*128
    float* coord_agg = msg_agg + (size_t)NN * NH;       // N*3
    bf16_t* We1p = (bf16_t*)(coord_agg + (size_t)NN * 3);   // 9*8*512
    bf16_t* We2p = We1p + 9 * 8 * 512;                      // 4*8*512
    bf16_t* Wx1p = We2p + 4 * 8 * 512;
    bf16_t* Wh1p = Wx1p + 4 * 8 * 512;                      // 8*8*512
    bf16_t* Wh2p = Wh1p + 8 * 8 * 512;

    // zero accumulators
    {
        int n = NN * (NH + 3);
        zero_f32<<<(n + 255) / 256, 256, 0, stream>>>(msg_agg, n);
    }
    // pack weights to bf16 WMMA-B layout
    pack_b<<<(9 * 8 * 512 + 255) / 256, 256, 0, stream>>>(We1, We1p, 273, 128, 9, 8);
    pack_b<<<(4 * 8 * 512 + 255) / 256, 256, 0, stream>>>(We2, We2p, 128, 128, 4, 8);
    pack_b<<<(4 * 8 * 512 + 255) / 256, 256, 0, stream>>>(Wx1, Wx1p, 128, 128, 4, 8);
    pack_b<<<(8 * 8 * 512 + 255) / 256, 256, 0, stream>>>(Wh1, Wh1p, 256, 128, 8, 8);
    pack_b<<<(4 * 8 * 512 + 255) / 256, 256, 0, stream>>>(Wh2, Wh2p, 128, 128, 4, 8);

    // edge phase: 40000 tiles of 16 edges, 4 waves/block
    edge_kernel<<<NE / 16 / 4, 128, 0, stream>>>(
        h, x, src, dst, ea, We1p, be1, We2p, be2, Wx1p, bx1, Wx2, bx2,
        msg_agg, coord_agg);

    // node phase: 1250 tiles of 16 nodes, 2 waves/block
    node_kernel<<<NN / 16 / 2, 64, 0, stream>>>(
        h, x, fm, Wh1p, bh1, Wh2p, bh2, msg_agg, coord_agg,
        (float*)d_out, (float*)d_out + (size_t)NN * NH);
}
